// TransformerEncoderLayer_4329327034666
// MI455X (gfx1250) — matrix operand order repaired and checked by reading, not compiled, not run on hardware
//
#include <hip/hip_runtime.h>

// ---------------------------------------------------------------------------
// Types for CDNA5 WMMA (wave32): v_wmma_f32_16x16x32_bf16
// ---------------------------------------------------------------------------
typedef __bf16 v16bf __attribute__((ext_vector_type(16)));
typedef __bf16 v8bf  __attribute__((ext_vector_type(8)));
typedef float  v8f   __attribute__((ext_vector_type(8)));

#define Bsz    16
#define Lseq   144
#define Dm     512
#define Hh     8
#define DHh    64
#define DSs    64
#define LLp    (144 * 144)
#define Npairs (16 * 144 * 144)
#define DFFn   2048
#define Mrows  (16 * 144)   // 2304

// ---------------------------------------------------------------------------
// Async global->LDS copy helpers (CDNA5 Tensor/Async path, ASYNCcnt-tracked).
// Each active lane copies 16 bytes: LDS dest addr in a VGPR, global src in a
// VGPR pair. Completion via s_wait_asynccnt then workgroup barrier.
// ---------------------------------------------------------------------------
__device__ __forceinline__ void async_copy_b128(void* lds_dst, const void* gsrc) {
  unsigned l = (unsigned)(size_t)lds_dst;
  unsigned long long g = (unsigned long long)(size_t)gsrc;
  asm volatile("global_load_async_to_lds_b128 %0, %1, off"
               :: "v"(l), "v"(g) : "memory");
}
__device__ __forceinline__ void async_wait0() {
  asm volatile("s_wait_asynccnt 0x0" ::: "memory");
}

// ---------------------------------------------------------------------------
// Fragment helpers.
// A fragment (16x32 bf16, row-major source, row stride must be 16B aligned):
//   lane l: m = l&15, khalf = l>>4
//   elems 0..15 -> K = k0 + khalf*16 + i       (32B contiguous)
// B fragment from transposed B (BT[n][K], 16 contiguous K per lane):
//   lane l: n = l&15, khalf = l>>4, K = k0 + khalf*16 + i
// C tile: lane l: n = l&15; row r -> m = (l>>4)*8 + r
// ---------------------------------------------------------------------------
__device__ __forceinline__ v16bf frag_a(const __bf16* rowk0, int khalf) {
  v8bf lo = *(const v8bf*)(rowk0 + khalf * 16);
  v8bf hi = *(const v8bf*)(rowk0 + khalf * 16 + 8);
  v16bf f;
#pragma unroll
  for (int i = 0; i < 8; i++) { f[i] = lo[i]; f[i + 8] = hi[i]; }
  return f;
}

__device__ __forceinline__ v16bf frag_bT(const __bf16* rowk0, int khalf) {
  v8bf lo = *(const v8bf*)(rowk0 + khalf * 16);
  v8bf hi = *(const v8bf*)(rowk0 + khalf * 16 + 8);
  v16bf f;
#pragma unroll
  for (int i = 0; i < 8; i++) { f[i] = lo[i]; f[i + 8] = hi[i]; }
  return f;
}

__device__ __forceinline__ v8f wmma_bf(v16bf a, v16bf b, v8f c) {
  return __builtin_amdgcn_wmma_f32_16x16x32_bf16(false, a, false, b,
                                                 (short)0, c, false, false);
}

// ---------------------------------------------------------------------------
// Weight prep: f32 [K][N] row-major -> bf16 [N][K] (BT layout for B-frags)
// ---------------------------------------------------------------------------
__global__ __launch_bounds__(256) void transpose_bf_k(
    const float* __restrict__ in, __bf16* __restrict__ out, int K, int Nt) {
  size_t total = (size_t)K * Nt;
  for (size_t idx = (size_t)blockIdx.x * blockDim.x + threadIdx.x; idx < total;
       idx += (size_t)gridDim.x * blockDim.x) {
    size_t n = idx / K, k = idx % K;
    out[idx] = (__bf16)in[k * (size_t)Nt + n];
  }
}

// WcB[o][t*64+c] = Wc[o][c][t]  (conv as 16x256x64 GEMM, B pre-transposed)
__global__ __launch_bounds__(256) void build_wcb_k(
    const float* __restrict__ Wc, __bf16* __restrict__ out) {
  int idx = blockIdx.x * blockDim.x + threadIdx.x;
  if (idx < 64 * 256) {
    int o = idx / 256, r = idx % 256;
    int t = r / 64, c = r % 64;
    out[idx] = (__bf16)Wc[((size_t)o * 64 + c) * 4 + t];
  }
}

__global__ __launch_bounds__(256) void f32_to_bf16_k(
    const float* __restrict__ in, __bf16* __restrict__ out, size_t n) {
  for (size_t i = (size_t)blockIdx.x * blockDim.x + threadIdx.x; i < n;
       i += (size_t)gridDim.x * blockDim.x)
    out[i] = (__bf16)in[i];
}

// ---------------------------------------------------------------------------
// LayerNorm over D=512, one block (256 threads) per row, bf16 output
// ---------------------------------------------------------------------------
__global__ __launch_bounds__(256) void ln512_k(const float* __restrict__ x,
                                               const float* __restrict__ g,
                                               const float* __restrict__ b,
                                               __bf16* __restrict__ out) {
  int row = blockIdx.x;
  const float* xr = x + (size_t)row * Dm;
  int tid = threadIdx.x;
  float v0 = xr[tid], v1 = xr[tid + 256];
  float s = v0 + v1;
#pragma unroll
  for (int o = 16; o > 0; o >>= 1) s += __shfl_xor(s, o, 32);
  __shared__ float red[8];
  if ((tid & 31) == 0) red[tid >> 5] = s;
  __syncthreads();
  float tot = 0.f;
#pragma unroll
  for (int i = 0; i < 8; i++) tot += red[i];
  float mean = tot * (1.0f / 512.0f);
  float d0 = v0 - mean, d1 = v1 - mean;
  float vs = d0 * d0 + d1 * d1;
#pragma unroll
  for (int o = 16; o > 0; o >>= 1) vs += __shfl_xor(vs, o, 32);
  __syncthreads();
  if ((tid & 31) == 0) red[tid >> 5] = vs;
  __syncthreads();
  float vt = 0.f;
#pragma unroll
  for (int i = 0; i < 8; i++) vt += red[i];
  float inv = rsqrtf(vt * (1.0f / 512.0f) + 1e-6f);
  out[(size_t)row * Dm + tid]       = (__bf16)(d0 * inv * g[tid] + b[tid]);
  out[(size_t)row * Dm + tid + 256] = (__bf16)(d1 * inv * g[tid + 256] + b[tid + 256]);
}

// ---------------------------------------------------------------------------
// Fused structure pipeline: 16 pairs per block (64 rows of 64).
//   async-DMA s tile (f32) to LDS -> bf16 -> s@sWq/k/v (+bias, q scaled)
//   -> 4-token/8-head attention -> ctx@sWo+b -> conv(16x256x64 WMMA)+ReLU
//   -> LN(64) -> R bf16.  Reads the 340MB structure tensor exactly once.
// The f32 staging buffer is union-overlaid with the q/k bf16 tiles (used in
// a later phase) to keep LDS usage flat.
// ---------------------------------------------------------------------------
__global__ __launch_bounds__(256) void structure_fused_k(
    const float* __restrict__ structure,  // [LL][4][16][64]
    const __bf16* __restrict__ sWqT, const __bf16* __restrict__ sWkT,
    const __bf16* __restrict__ sWvT, const __bf16* __restrict__ sWoT,
    const float* __restrict__ sbq, const float* __restrict__ sbk,
    const float* __restrict__ sbv, const float* __restrict__ sbo,
    const __bf16* __restrict__ WcB,  // [64][256]
    const float* __restrict__ bc, const float* __restrict__ gs,
    const float* __restrict__ bs,
    __bf16* __restrict__ Rout)  // [Npairs][64]
{
  __shared__ union {
    float f32[64][64];                                   // async-DMA staging
    struct { __bf16 q[64][72]; __bf16 k[64][72]; } qk;   // later phase
  } u;
  __shared__ __bf16 sS[64][72];      // s tile (bf16); reused as ctx
  __shared__ __bf16 sV[64][72];
  __shared__ __bf16 saF[16][264];    // sa flattened per pair: [pair][t*64+c]
  __shared__ float  convO[16][68];

  auto& sQ = u.qk.q;
  auto& sK = u.qk.k;

  int tid = threadIdx.x;
  int n0 = blockIdx.x * 16;  // first pair of this block

  // --- async-DMA gather of s tile (transpose (2,0,1,3)) into LDS f32 ---
  {
    int r = tid >> 2;               // 0..63  (row = lp*4 + t)
    int ch = tid & 3;               // 16B chunk within the 256B row
    int lp = r >> 2, t = r & 3;
    int n = n0 + lp;
    int b = n / LLp, pp = n % LLp;
    const float* src = structure + (((size_t)pp * 4 + t) * 16 + b) * 64 + ch * 4;
    async_copy_b128(&u.f32[r][ch * 4], src);
    async_wait0();
  }
  __syncthreads();
  // convert f32 -> bf16 into sS
  for (int e = tid; e < 64 * 64; e += 256) {
    int r = e >> 6, c = e & 63;
    sS[r][c] = (__bf16)u.f32[r][c];
  }
  __syncthreads();

  int wave = tid >> 5, lane = tid & 31, nl = lane & 15, khalf = lane >> 4;

  // --- q,k,v GEMMs: 3 weights x 16 tiles (64x64, K=64) ---
  for (int job = wave; job < 48; job += 8) {
    int w = job / 16, tile = job % 16;
    int mi = tile / 4, ni = tile % 4;
    const __bf16* BT = (w == 0) ? sWqT : (w == 1) ? sWkT : sWvT;
    const float* bias = (w == 0) ? sbq : (w == 1) ? sbk : sbv;
    v8f acc = {};
#pragma unroll
    for (int kk = 0; kk < 64; kk += 32) {
      v16bf af = frag_a(&sS[mi * 16 + nl][kk], khalf);
      v16bf bf = frag_bT(BT + (size_t)(ni * 16 + nl) * 64 + kk, khalf);
      acc = wmma_bf(af, bf, acc);
    }
    __bf16 (*dst)[72] = (w == 0) ? sQ : (w == 1) ? sK : sV;
    float scl = (w == 0) ? 0.35355339059327373f : 1.0f;  // 1/sqrt(8) for q
#pragma unroll
    for (int r = 0; r < 8; r++) {
      int m = mi * 16 + khalf * 8 + r;
      int c = ni * 16 + nl;
      dst[m][c] = (__bf16)((acc[r] + bias[c]) * scl);
    }
  }
  __syncthreads();

  // --- tiny attention: 128 units = (pair, head); ctx overwrites sS ---
  if (tid < 128) {
    int lp = tid >> 3, h = tid & 7;
    float qv[4][8], kv[4][8], vv[4][8];
#pragma unroll
    for (int i = 0; i < 4; i++)
#pragma unroll
      for (int d = 0; d < 8; d++) {
        qv[i][d] = (float)sQ[lp * 4 + i][h * 8 + d];
        kv[i][d] = (float)sK[lp * 4 + i][h * 8 + d];
        vv[i][d] = (float)sV[lp * 4 + i][h * 8 + d];
      }
    float sc[4][4];
#pragma unroll
    for (int i = 0; i < 4; i++) {
      float mx = -3e38f;
#pragma unroll
      for (int j = 0; j < 4; j++) {
        float s = 0.f;
#pragma unroll
        for (int d = 0; d < 8; d++) s += qv[i][d] * kv[j][d];
        sc[i][j] = s; mx = fmaxf(mx, s);
      }
      float sum = 0.f;
#pragma unroll
      for (int j = 0; j < 4; j++) { sc[i][j] = __expf(sc[i][j] - mx); sum += sc[i][j]; }
      float inv = 1.f / sum;
#pragma unroll
      for (int j = 0; j < 4; j++) sc[i][j] *= inv;
    }
#pragma unroll
    for (int i = 0; i < 4; i++)
#pragma unroll
      for (int d = 0; d < 8; d++) {
        float s = 0.f;
#pragma unroll
        for (int j = 0; j < 4; j++) s += sc[i][j] * vv[j][d];
        sS[lp * 4 + i][h * 8 + d] = (__bf16)s;
      }
  }
  __syncthreads();

  // --- sa = ctx @ sWo + sbo -> saF[pair][t*64+c] ---
  for (int tile = wave; tile < 16; tile += 8) {
    int mi = tile / 4, ni = tile % 4;
    v8f acc = {};
#pragma unroll
    for (int kk = 0; kk < 64; kk += 32) {
      v16bf af = frag_a(&sS[mi * 16 + nl][kk], khalf);
      v16bf bf = frag_bT(sWoT + (size_t)(ni * 16 + nl) * 64 + kk, khalf);
      acc = wmma_bf(af, bf, acc);
    }
#pragma unroll
    for (int r = 0; r < 8; r++) {
      int m = mi * 16 + khalf * 8 + r;
      int c = ni * 16 + nl;
      saF[m >> 2][(m & 3) * 64 + c] = (__bf16)(acc[r] + sbo[c]);
    }
  }
  __syncthreads();

  // --- conv GEMM: M=16 pairs, N=64, K=256; +bias, ReLU ---
  for (int ni = wave; ni < 4; ni += 8) {
    v8f acc = {};
#pragma unroll
    for (int k0 = 0; k0 < 256; k0 += 32) {
      v16bf af = frag_a(&saF[nl][k0], khalf);
      v16bf bf = frag_bT(WcB + (size_t)(ni * 16 + nl) * 256 + k0, khalf);
      acc = wmma_bf(af, bf, acc);
    }
#pragma unroll
    for (int r = 0; r < 8; r++) {
      int m = khalf * 8 + r;
      int c = ni * 16 + nl;
      convO[m][c] = fmaxf(acc[r] + bc[c], 0.f);
    }
  }
  __syncthreads();

  // --- LN over 64 per pair, write R bf16 ---
  if (tid < 16) {
    int lp = tid;
    float mean = 0.f;
    for (int c = 0; c < 64; c++) mean += convO[lp][c];
    mean *= (1.f / 64.f);
    float var = 0.f;
    for (int c = 0; c < 64; c++) { float d = convO[lp][c] - mean; var += d * d; }
    var *= (1.f / 64.f);
    float inv = rsqrtf(var + 1e-6f);
    size_t base = ((size_t)n0 + lp) * 64;
    for (int c = 0; c < 64; c++)
      Rout[base + c] = (__bf16)((convO[lp][c] - mean) * inv * gs[c] + bs[c]);
  }
}

// ---------------------------------------------------------------------------
// Generic bf16 WMMA GEMM: 64x64 tile per 256-thread block. A-tiles are
// async-DMA'd global->LDS (no VGPR round trip; co-executes with WMMA);
// B-frags read from pre-transposed global BT[N][K] (L2-resident weights).
// Next A k-chunk is prefetched (global_prefetch_b8). Epilogue: bias, scale
// ((acc+bias)*scale), relu, residual, f32/bf16 output, optional
// [B][H][L][DH] scatter for q/k/v.
// ---------------------------------------------------------------------------
__global__ __launch_bounds__(256) void gemm_bf16_k(
    const __bf16* __restrict__ A, int lda,
    const __bf16* __restrict__ BT, int ldb,
    const float* __restrict__ bias,
    const float* __restrict__ resid,
    float* __restrict__ outF,
    __bf16* __restrict__ outB,
    int M, int Nt, int K,
    float scale, int relu, int qkvmap) {
  __shared__ __bf16 sA[64][72];
  int tn = Nt >> 6;
  int bm = blockIdx.x / tn;
  int bn = blockIdx.x % tn;
  int tid = threadIdx.x;
  int wave = tid >> 5, lane = tid & 31, nl = lane & 15, khalf = lane >> 4;
  int mi = wave >> 1;
  int ni0 = (wave & 1) * 2;
  v8f acc0 = {}; v8f acc1 = {};

  for (int k0 = 0; k0 < K; k0 += 64) {
    int r = tid >> 2, c = (tid & 3) * 16;
    const __bf16* src = A + (size_t)(bm * 64 + r) * lda + k0 + c;
    async_copy_b128(&sA[r][c], src);
    async_copy_b128(&sA[r][c + 8], src + 8);
    if (k0 + 64 < K) __builtin_prefetch(src + 64, 0, 3);
    async_wait0();
    __syncthreads();
    const __bf16* bt0 = BT + (size_t)(bn * 64 + ni0 * 16 + nl) * ldb + k0;
    const __bf16* bt1 = BT + (size_t)(bn * 64 + (ni0 + 1) * 16 + nl) * ldb + k0;
#pragma unroll
    for (int kk = 0; kk < 64; kk += 32) {
      v16bf af = frag_a(&sA[mi * 16 + nl][kk], khalf);
      acc0 = wmma_bf(af, frag_bT(bt0 + kk, khalf), acc0);
      acc1 = wmma_bf(af, frag_bT(bt1 + kk, khalf), acc1);
    }
    __syncthreads();
  }

#pragma unroll
  for (int r = 0; r < 8; r++) {
    int mg = bm * 64 + mi * 16 + khalf * 8 + r;
#pragma unroll
    for (int t = 0; t < 2; t++) {
      int ng = bn * 64 + (ni0 + t) * 16 + nl;
      float v = t ? acc1[r] : acc0[r];
      if (bias) v += bias[ng];
      v *= scale;
      if (relu) v = fmaxf(v, 0.f);
      if (resid) v += resid[(size_t)mg * Nt + ng];
      if (qkvmap) {
        int b = mg / Lseq, l = mg % Lseq, h = ng >> 6, d = ng & 63;
        outB[(((size_t)b * Hh + h) * Lseq + l) * 64 + d] = (__bf16)v;
      } else {
        if (outF) outF[(size_t)mg * Nt + ng] = v;
        if (outB) outB[(size_t)mg * Nt + ng] = (__bf16)v;
      }
    }
  }
}

// ---------------------------------------------------------------------------
// scores = q @ k^T per (b,h): 144x144, K=64.  q,k async-DMA'd into LDS.
// ---------------------------------------------------------------------------
__global__ __launch_bounds__(256) void scores_qk_k(
    const __bf16* __restrict__ q, const __bf16* __restrict__ k,
    float* __restrict__ scores) {
  int bh = blockIdx.x;
  const __bf16* qh = q + (size_t)bh * Lseq * 64;
  const __bf16* kh = k + (size_t)bh * Lseq * 64;
  __shared__ __bf16 sQ[144][72];
  __shared__ __bf16 sK[144][72];
  int tid = threadIdx.x;
  for (int ch = tid; ch < 1152; ch += 256) {
    int r = ch >> 3, c = (ch & 7) * 8;
    async_copy_b128(&sQ[r][c], qh + (size_t)r * 64 + c);
    async_copy_b128(&sK[r][c], kh + (size_t)r * 64 + c);
  }
  async_wait0();
  __syncthreads();
  int wave = tid >> 5, lane = tid & 31, nl = lane & 15, khalf = lane >> 4;
  float* out = scores + (size_t)bh * Lseq * Lseq;
  for (int tile = wave; tile < 81; tile += 8) {
    int mi = tile / 9, ni = tile % 9;
    v8f acc = {};
#pragma unroll
    for (int kk = 0; kk < 64; kk += 32) {
      v16bf af = frag_a(&sQ[mi * 16 + nl][kk], khalf);
      v16bf bf = frag_bT(&sK[ni * 16 + nl][kk], khalf);
      acc = wmma_bf(af, bf, acc);
    }
#pragma unroll
    for (int r = 0; r < 8; r++) {
      int m = mi * 16 + khalf * 8 + r;
      int n = ni * 16 + nl;
      out[(size_t)m * Lseq + n] = acc[r];
    }
  }
}

// ---------------------------------------------------------------------------
// scores[b,:,i,:] += q[b,:,i,:] @ R[b,i,:,:]^T  per (b,i). M=8(h) padded to 16.
// ---------------------------------------------------------------------------
__global__ __launch_bounds__(256) void scores_qr_k(
    const __bf16* __restrict__ q, const __bf16* __restrict__ Rb,
    float* __restrict__ scores) {
  int b = blockIdx.x / Lseq, i = blockIdx.x % Lseq;
  __shared__ __bf16 sQ[16][72];
  int tid = threadIdx.x;
  if (tid < 128) {
    int r = tid >> 3, c = (tid & 7) * 8;
    if (r < 8) {
      const __bf16* src = q + (((size_t)b * Hh + r) * Lseq + i) * 64 + c;
      *(v8bf*)&sQ[r][c] = *(const v8bf*)src;
    } else {
      v8bf z = {};
      *(v8bf*)&sQ[r][c] = z;
    }
  }
  __syncthreads();
  int wave = tid >> 5, lane = tid & 31, nl = lane & 15, khalf = lane >> 4;
  const __bf16* Rrow = Rb + ((size_t)b * LLp + (size_t)i * Lseq) * 64;
  for (int ni = wave; ni < 9; ni += 8) {
    v8f acc = {};
#pragma unroll
    for (int kk = 0; kk < 64; kk += 32) {
      v16bf af = frag_a(&sQ[nl][kk], khalf);
      v16bf bf = frag_bT(Rrow + (size_t)(ni * 16 + nl) * 64 + kk, khalf);
      acc = wmma_bf(af, bf, acc);
    }
#pragma unroll
    for (int r = 0; r < 8; r++) {
      int h = khalf * 8 + r;
      if (h < 8) {
        int j = ni * 16 + nl;
        float* p = scores + (((size_t)(b * Hh + h)) * Lseq + i) * Lseq + j;
        *p += acc[r];
      }
    }
  }
}

// ---------------------------------------------------------------------------
// mask + softmax over j (144), write attn bf16 padded to K=160 (zeros).
// One wave per row; 8 rows per block.
// ---------------------------------------------------------------------------
__global__ __launch_bounds__(256) void softmax_mask_k(
    const float* __restrict__ scores, const int* __restrict__ mask,
    __bf16* __restrict__ attn) {
  int row = blockIdx.x * 8 + (threadIdx.x >> 5);  // row in [0, B*H*L)
  int lane = threadIdx.x & 31;
  int i = row % Lseq;
  int bh = row / Lseq;
  int b = bh / Hh;
  const float* sr = scores + (size_t)row * Lseq;
  const int* mr = mask + ((size_t)b * Lseq + i) * Lseq;
  float vals[5];
  float mx = -3.0e38f;
#pragma unroll
  for (int t = 0; t < 5; t++) {
    int j = lane + 32 * t;
    float v = -3.0e38f;
    if (j < Lseq) { v = sr[j]; if (mr[j] != 0) v = -1e18f; }
    vals[t] = v; mx = fmaxf(mx, v);
  }
#pragma unroll
  for (int o = 16; o > 0; o >>= 1) mx = fmaxf(mx, __shfl_xor(mx, o, 32));
  float sum = 0.f;
#pragma unroll
  for (int t = 0; t < 5; t++) {
    int j = lane + 32 * t;
    float e = (j < Lseq) ? __expf(vals[t] - mx) : 0.f;
    vals[t] = e; sum += e;
  }
#pragma unroll
  for (int o = 16; o > 0; o >>= 1) sum += __shfl_xor(sum, o, 32);
  float inv = 1.0f / sum;
  __bf16* ar = attn + (size_t)row * 160;
#pragma unroll
  for (int t = 0; t < 5; t++) {
    int j = lane + 32 * t;
    ar[j] = (j < Lseq) ? (__bf16)(vals[t] * inv) : (__bf16)0.f;
  }
}

// ---------------------------------------------------------------------------
// ctx = attn @ v per (b,h): 144x64, K=160 (padded). v transposed into LDS.
// Writes ctx f32 in [B][L][D] layout at column h*64+d.
// ---------------------------------------------------------------------------
__global__ __launch_bounds__(256) void ctx_av_k(
    const __bf16* __restrict__ attn, const __bf16* __restrict__ v,
    float* __restrict__ ctx) {
  int bh = blockIdx.x;
  int b = bh / Hh, h = bh % Hh;
  __shared__ __bf16 sVT[64][168];
  int tid = threadIdx.x;
  for (int e = tid; e < 64 * 160; e += 256) {
    int d = e / 160, j = e % 160;
    __bf16 bv = (__bf16)0.f;
    if (j < Lseq) bv = v[((size_t)bh * Lseq + j) * 64 + d];
    sVT[d][j] = bv;
  }
  __syncthreads();
  const __bf16* arow = attn + (size_t)bh * Lseq * 160;
  int wave = tid >> 5, lane = tid & 31, nl = lane & 15, khalf = lane >> 4;
  for (int tile = wave; tile < 36; tile += 8) {  // 9 m-tiles x 4 n-tiles
    int mi = tile / 4, ni = tile % 4;
    v8f acc = {};
#pragma unroll
    for (int k0 = 0; k0 < 160; k0 += 32) {
      v16bf af = frag_a(arow + (size_t)(mi * 16 + nl) * 160 + k0, khalf);
      v16bf bf = frag_bT(&sVT[ni * 16 + nl][k0], khalf);
      acc = wmma_bf(af, bf, acc);
    }
#pragma unroll
    for (int r = 0; r < 8; r++) {
      int l = mi * 16 + khalf * 8 + r;
      int d = ni * 16 + nl;
      ctx[((size_t)b * Lseq + l) * Dm + h * 64 + d] = acc[r];
    }
  }
}

// ---------------------------------------------------------------------------
// ctx[b,i,h*64+d] += sum_j attn[b,h,i,j] * R[b,i,j,d]  per (b,i).
// M=8(h) padded to 16, K=160 (padded), N=64.
// ---------------------------------------------------------------------------
__global__ __launch_bounds__(256) void ctx_ar_k(
    const __bf16* __restrict__ attn, const __bf16* __restrict__ Rb,
    float* __restrict__ ctx) {
  int b = blockIdx.x / Lseq, i = blockIdx.x % Lseq;
  __shared__ __bf16 sRT[64][168];
  __shared__ __bf16 sA[16][168];
  int tid = threadIdx.x;
  for (int e = tid; e < 64 * 160; e += 256) {
    int d = e / 160, j = e % 160;
    __bf16 bv = (__bf16)0.f;
    if (j < Lseq) bv = Rb[((size_t)b * LLp + (size_t)i * Lseq + j) * 64 + d];
    sRT[d][j] = bv;
  }
  for (int e = tid; e < 16 * 160; e += 256) {
    int r = e / 160, j = e % 160;
    __bf16 bv = (__bf16)0.f;
    if (r < 8) bv = attn[(((size_t)(b * Hh + r)) * Lseq + i) * 160 + j];
    sA[r][j] = bv;
  }
  __syncthreads();
  int wave = tid >> 5, lane = tid & 31, nl = lane & 15, khalf = lane >> 4;
  for (int ni = wave; ni < 4; ni += 8) {
    v8f acc = {};
#pragma unroll
    for (int k0 = 0; k0 < 160; k0 += 32) {
      v16bf af = frag_a(&sA[nl][k0], khalf);
      v16bf bf = frag_bT(&sRT[ni * 16 + nl][k0], khalf);
      acc = wmma_bf(af, bf, acc);
    }
#pragma unroll
    for (int r = 0; r < 8; r++) {
      int h = khalf * 8 + r;
      if (h < 8) {
        int d = ni * 16 + nl;
        ctx[((size_t)b * Lseq + i) * Dm + h * 64 + d] += acc[r];
      }
    }
  }
}

// ---------------------------------------------------------------------------
// Launch
// ---------------------------------------------------------------------------
extern "C" void kernel_launch(void* const* d_in, const int* in_sizes, int n_in,
                              void* d_out, int out_size, void* d_ws,
                              size_t ws_size, hipStream_t stream) {
  (void)in_sizes; (void)n_in; (void)out_size; (void)ws_size;
  const float* inputs    = (const float*)d_in[0];
  const float* structure = (const float*)d_in[1];
  const int*   mask      = (const int*)d_in[2];
  const float* g_att = (const float*)d_in[3];
  const float* b_att = (const float*)d_in[4];
  const float* Wq = (const float*)d_in[5];   const float* bq = (const float*)d_in[6];
  const float* Wk = (const float*)d_in[7];   const float* bk = (const float*)d_in[8];
  const float* Wv = (const float*)d_in[9];   const float* bv = (const float*)d_in[10];
  const float* Wo = (const float*)d_in[11];  const float* bo = (const float*)d_in[12];
  const float* sWq = (const float*)d_in[13]; const float* sbq = (const float*)d_in[14];
  const float* sWk = (const float*)d_in[15]; const float* sbk = (const float*)d_in[16];
  const float* sWv = (const float*)d_in[17]; const float* sbv = (const float*)d_in[18];
  const float* sWo = (const float*)d_in[19]; const float* sbo = (const float*)d_in[20];
  const float* Wc = (const float*)d_in[21];  const float* bc = (const float*)d_in[22];
  const float* g_s = (const float*)d_in[23]; const float* b_s = (const float*)d_in[24];
  const float* g_ffn = (const float*)d_in[25]; const float* b_ffn = (const float*)d_in[26];
  const float* W1 = (const float*)d_in[27];  const float* b1 = (const float*)d_in[28];
  const float* W2 = (const float*)d_in[29];  const float* b2 = (const float*)d_in[30];
  float* out = (float*)d_out;

  char* p = (char*)d_ws;
  auto take = [&](size_t elems, size_t esz) {
    void* q = (void*)p;
    p += (elems * esz + 255) & ~(size_t)255;
    return q;
  };
  __bf16* R_bf   = (__bf16*)take((size_t)Npairs * 64, 2);
  __bf16* xn_bf  = (__bf16*)take((size_t)Mrows * Dm, 2);
  __bf16* q_bf   = (__bf16*)take((size_t)Mrows * Dm, 2);
  __bf16* k_bf   = (__bf16*)take((size_t)Mrows * Dm, 2);
  __bf16* v_bf   = (__bf16*)take((size_t)Mrows * Dm, 2);
  float*  scores = (float*) take((size_t)Bsz * Hh * Lseq * Lseq, 4);
  __bf16* attn   = (__bf16*)take((size_t)Bsz * Hh * Lseq * 160, 2);
  float*  ctxF   = (float*) take((size_t)Mrows * Dm, 4);
  __bf16* ctx_bf = (__bf16*)take((size_t)Mrows * Dm, 2);
  float*  x_ws   = (float*) take((size_t)Mrows * Dm, 4);
  __bf16* xn2    = (__bf16*)take((size_t)Mrows * Dm, 2);
  __bf16* h1     = (__bf16*)take((size_t)Mrows * DFFn, 2);
  __bf16* WqT  = (__bf16*)take(512 * 512, 2);
  __bf16* WkT  = (__bf16*)take(512 * 512, 2);
  __bf16* WvT  = (__bf16*)take(512 * 512, 2);
  __bf16* WoT  = (__bf16*)take(512 * 512, 2);
  __bf16* W1T  = (__bf16*)take(512 * 2048, 2);
  __bf16* W2T  = (__bf16*)take(512 * 2048, 2);
  __bf16* sWqT = (__bf16*)take(64 * 64, 2);
  __bf16* sWkT = (__bf16*)take(64 * 64, 2);
  __bf16* sWvT = (__bf16*)take(64 * 64, 2);
  __bf16* sWoT = (__bf16*)take(64 * 64, 2);
  __bf16* WcB  = (__bf16*)take(64 * 256, 2);

  // weight prep
  transpose_bf_k<<<1024, 256, 0, stream>>>(Wq, WqT, 512, 512);
  transpose_bf_k<<<1024, 256, 0, stream>>>(Wk, WkT, 512, 512);
  transpose_bf_k<<<1024, 256, 0, stream>>>(Wv, WvT, 512, 512);
  transpose_bf_k<<<1024, 256, 0, stream>>>(Wo, WoT, 512, 512);
  transpose_bf_k<<<2048, 256, 0, stream>>>(W1, W1T, 512, 2048);   // -> [2048][512]
  transpose_bf_k<<<2048, 256, 0, stream>>>(W2, W2T, 2048, 512);   // -> [512][2048]
  transpose_bf_k<<<16,   256, 0, stream>>>(sWq, sWqT, 64, 64);
  transpose_bf_k<<<16,   256, 0, stream>>>(sWk, sWkT, 64, 64);
  transpose_bf_k<<<16,   256, 0, stream>>>(sWv, sWvT, 64, 64);
  transpose_bf_k<<<16,   256, 0, stream>>>(sWo, sWoT, 64, 64);
  build_wcb_k<<<64, 256, 0, stream>>>(Wc, WcB);

  // LN(inputs)
  ln512_k<<<Mrows, 256, 0, stream>>>(inputs, g_att, b_att, xn_bf);

  // fused structure pipeline -> R
  structure_fused_k<<<Npairs / 16, 256, 0, stream>>>(
      structure, sWqT, sWkT, sWvT, sWoT, sbq, sbk, sbv, sbo, WcB, bc, g_s, b_s,
      R_bf);

  // main q,k,v projections (q scaled by 1/sqrt(64))
  gemm_bf16_k<<<36 * 8, 256, 0, stream>>>(xn_bf, 512, WqT, 512, bq, nullptr,
                                          nullptr, q_bf, Mrows, 512, 512,
                                          0.125f, 0, 1);
  gemm_bf16_k<<<36 * 8, 256, 0, stream>>>(xn_bf, 512, WkT, 512, bk, nullptr,
                                          nullptr, k_bf, Mrows, 512, 512, 1.f,
                                          0, 1);
  gemm_bf16_k<<<36 * 8, 256, 0, stream>>>(xn_bf, 512, WvT, 512, bv, nullptr,
                                          nullptr, v_bf, Mrows, 512, 512, 1.f,
                                          0, 1);

  // scores = q@k^T + q@R^T, mask, softmax
  scores_qk_k<<<Bsz * Hh, 256, 0, stream>>>(q_bf, k_bf, scores);
  scores_qr_k<<<Bsz * Lseq, 256, 0, stream>>>(q_bf, R_bf, scores);
  softmax_mask_k<<<Bsz * Hh * Lseq / 8, 256, 0, stream>>>(scores, mask, attn);

  // ctx = attn@v + attn@R
  ctx_av_k<<<Bsz * Hh, 256, 0, stream>>>(attn, v_bf, ctxF);
  ctx_ar_k<<<Bsz * Lseq, 256, 0, stream>>>(attn, R_bf, ctxF);
  f32_to_bf16_k<<<2048, 256, 0, stream>>>(ctxF, ctx_bf, (size_t)Mrows * Dm);

  // x = ctx@Wo + bo + inputs
  gemm_bf16_k<<<36 * 8, 256, 0, stream>>>(ctx_bf, 512, WoT, 512, bo, inputs,
                                          x_ws, nullptr, Mrows, 512, 512, 1.f,
                                          0, 0);

  // FFN: h = relu(LN(x)@W1 + b1); out = h@W2 + b2 + x
  ln512_k<<<Mrows, 256, 0, stream>>>(x_ws, g_ffn, b_ffn, xn2);
  gemm_bf16_k<<<36 * 32, 256, 0, stream>>>(xn2, 512, W1T, 512, b1, nullptr,
                                           nullptr, h1, Mrows, 2048, 512, 1.f,
                                           1, 0);
  gemm_bf16_k<<<36 * 8, 256, 0, stream>>>(h1, 2048, W2T, 2048, b2, x_ws, out,
                                          nullptr, Mrows, 512, 2048, 1.f, 0, 0);
}
